// MultiScaleSTN_7249904795800
// MI455X (gfx1250) — matrix operand order repaired and checked by reading, not compiled, not run on hardware
//
#include <hip/hip_runtime.h>

// ---------------------------------------------------------------------------
// MultiScaleSTN for MI455X (gfx1250, wave32, WMMA 16x16x32 bf16)
// All intermediates stored transposed [spatial][channel] so every GEMM tile
// stage is a contiguous 16B chunk copy -> async global->LDS eligible.
// ---------------------------------------------------------------------------

typedef __attribute__((ext_vector_type(16))) __bf16 v16bf;
typedef __attribute__((ext_vector_type(8)))  float  v8f;
typedef __attribute__((ext_vector_type(4)))  int    v4i;

#define LEAKY_SLOPE 0.1f
#define BN_EPS      1e-5f

#define AS1 __attribute__((address_space(1)))
#define AS3 __attribute__((address_space(3)))

#if defined(__HIP_DEVICE_COMPILE__) && \
    __has_builtin(__builtin_amdgcn_global_load_async_to_lds_b128) && \
    __has_builtin(__builtin_amdgcn_s_wait_asynccnt)
#define USE_ASYNC_LDS 1
#else
#define USE_ASYNC_LDS 0
#endif

union FragBF {
    float4 f4[2];
    v16bf  v;
};

// 16-byte global -> LDS chunk copy (async on CDNA5 when available)
__device__ __forceinline__ void copy16(const __bf16* g, __bf16* l) {
#if USE_ASYNC_LDS
    __builtin_amdgcn_global_load_async_to_lds_b128(
        (AS1 v4i*)g, (AS3 v4i*)l, 0, 0);
#else
    *(uint4*)l = *(const uint4*)g;
#endif
}

__device__ __forceinline__ void stage_fence() {
#if USE_ASYNC_LDS
    __builtin_amdgcn_s_wait_asynccnt(0);
#endif
}

// ---------------------------------------------------------------------------
// f32 -> bf16 conversion (grid-stride), for w0 (layout already GEMM-friendly)
// ---------------------------------------------------------------------------
__global__ void cvt_f32_bf16(const float* __restrict__ in,
                             __bf16* __restrict__ out, long n) {
    long i = (long)blockIdx.x * blockDim.x + threadIdx.x;
    long stride = (long)gridDim.x * blockDim.x;
    for (; i < n; i += stride) out[i] = (__bf16)in[i];
}

// ---------------------------------------------------------------------------
// Transpose-convert: f32 [B][C][S] -> bf16 [B][S][C]  (LDS 32x32 tiles)
// grid (S/32, C/32, B), block 1024 (= 32x32)
// ---------------------------------------------------------------------------
__global__ __launch_bounds__(1024)
void transpose_cvt_f32_bf16(const float* __restrict__ in,
                            __bf16* __restrict__ out, int C, int S) {
    __shared__ __bf16 tile[32][33];
    int t  = threadIdx.x;
    int tx = t & 31, ty = t >> 5;
    int s0 = blockIdx.x * 32, c0 = blockIdx.y * 32, b = blockIdx.z;
    const float* ib = in  + (long)b * C * S;
    __bf16*      ob = out + (long)b * C * S;
    tile[tx][ty] = (__bf16)ib[(long)(c0 + ty) * S + (s0 + tx)];
    __syncthreads();
    ob[(long)(s0 + ty) * C + (c0 + tx)] = tile[ty][tx];
}

// ---------------------------------------------------------------------------
// Weight permute+convert for 3x3 convs:
//   in  f32 [Co][Ci][3][3]   (k_in  = ci*9 + tap)
//   out bf16 [Co][9*Ci]      (k_out = tap*Ci + ci)  -- tap-major K
// ---------------------------------------------------------------------------
__global__ void permute_w_cvt(const float* __restrict__ in,
                              __bf16* __restrict__ out, int Co, int Ci) {
    long total = (long)Co * Ci * 9;
    long i = (long)blockIdx.x * blockDim.x + threadIdx.x;
    long stride = (long)gridDim.x * blockDim.x;
    for (; i < total; i += stride) {
        long co  = i / ((long)Ci * 9);
        long rem = i - co * (long)Ci * 9;
        long tap = rem / Ci;
        long ci  = rem - tap * Ci;
        out[i] = (__bf16)in[co * (long)Ci * 9 + ci * 9 + tap];
    }
}

// ---------------------------------------------------------------------------
// WMMA GEMM: C[M,N] = sum_K A[M,K] * B[K,N], output stored TRANSPOSED [N][ldo]
//   A: bf16 row-major [M][K] (lda = K)
//   BGATHER 0: B given transposed, Bm = B^T row-major [N][ldb] (ldb >= K)
//   BGATHER 1: implicit 3x3 conv from X^T [1024][Ci] with tap-major K:
//              k = tap*Ci + ci; one BK=32 tile sits inside a single tap, so the
//              (dy,dx) shift is uniform per tile; border chunks are zero-filled.
//   EPI 1: fold BN (g,b,m,v) + leaky ReLU into the epilogue.
// Tiling: BM=BN=128, BK=32; 256 threads = 8 wave32 in 2(M) x 4(N) grid;
// each wave computes 64x32 = 4x2 WMMA 16x16x32 tiles. Double-buffered LDS,
// async staging of tile k+1 overlapped with WMMA on tile k.
// ---------------------------------------------------------------------------
template<int BGATHER, int EPI>
__global__ __launch_bounds__(256)
void gemm_wmma_kernel(const __bf16* __restrict__ A,
                      const __bf16* __restrict__ Bm,
                      __bf16* __restrict__ OutT,
                      int K, int lda, int ldb, int ldo, int Ci,
                      long a_bstride, long b_bstride, long o_bstride,
                      const float* __restrict__ bng, const float* __restrict__ bnb,
                      const float* __restrict__ bnm, const float* __restrict__ bnv)
{
    __shared__ __bf16 As[2][128][40];   // [m][k] padded (80B row stride)
    __shared__ __bf16 Bs[2][128][40];   // [n][k] padded

    const int tid  = threadIdx.x;
    const int lane = tid & 31;
    const int wid  = tid >> 5;
    const int wm   = wid & 1;           // 2 waves along M (64 rows each)
    const int wn   = wid >> 1;          // 4 waves along N (32 cols each)
    const int half = lane >> 4;
    const int l15  = lane & 15;

    const int Nbase = blockIdx.x * 128;
    const int Mbase = blockIdx.y * 128;
    const int batch = blockIdx.z;

    const __bf16* Ab = A  + (long)batch * a_bstride;
    const __bf16* Bb = Bm + (long)batch * b_bstride;
    __bf16* Ob = OutT + (long)batch * o_bstride;

    // ---- staging: 512 chunks of 16B each for A and B, 2 chunks per thread ----
    auto stageA = [&](int k0, int bf) {
        #pragma unroll
        for (int j = 0; j < 2; ++j) {
            int e   = tid + (j << 8);
            int row = e >> 2;
            int kc  = (e & 3) << 3;
            copy16(Ab + (long)(Mbase + row) * lda + (k0 + kc), &As[bf][row][kc]);
        }
    };
    auto stageB = [&](int k0, int bf) {
        if (BGATHER == 0) {
            #pragma unroll
            for (int j = 0; j < 2; ++j) {
                int e  = tid + (j << 8);
                int n  = e >> 2;
                int kc = (e & 3) << 3;
                copy16(Bb + (long)(Nbase + n) * ldb + (k0 + kc), &Bs[bf][n][kc]);
            }
        } else {
            int tap = k0 / Ci;                  // uniform over the BK tile
            int cb  = k0 - tap * Ci;
            int dy  = tap / 3 - 1;
            int dx  = tap - (tap / 3) * 3 - 1;
            #pragma unroll
            for (int j = 0; j < 2; ++j) {
                int e  = tid + (j << 8);
                int n  = e >> 2;
                int kc = (e & 3) << 3;
                int ng = Nbase + n;
                int x  = (ng & 31) + dx;
                int y  = (ng >> 5) + dy;
                if ((unsigned)x < 32u && (unsigned)y < 32u) {
                    copy16(Bb + (long)((y << 5) + x) * ldb + (cb + kc),
                           &Bs[bf][n][kc]);
                } else {
                    uint4 z = {0u, 0u, 0u, 0u};
                    *(uint4*)&Bs[bf][n][kc] = z;
                }
            }
        }
    };

    v8f acc[4][2] = {};

    const int ktiles = K >> 5;
    stageA(0, 0);
    stageB(0, 0);
    stage_fence();
    __syncthreads();

    for (int kt = 0; kt < ktiles; ++kt) {
        const int cur = kt & 1;
        if (kt + 1 < ktiles) {                 // overlap next tile with compute
            stageA((kt + 1) << 5, cur ^ 1);
            stageB((kt + 1) << 5, cur ^ 1);
        }

        FragBF af[4], bfr[2];
        #pragma unroll
        for (int mt = 0; mt < 4; ++mt) {
            int row = wm * 64 + mt * 16 + l15;
            int kb  = half << 3;               // 0 or 8
            af[mt].f4[0] = *(const float4*)&As[cur][row][kb];
            af[mt].f4[1] = *(const float4*)&As[cur][row][16 + kb];
        }
        #pragma unroll
        for (int nt = 0; nt < 2; ++nt) {
            int col = wn * 32 + nt * 16 + l15;
            int kb  = half << 3;               // 0 or 8
            bfr[nt].f4[0] = *(const float4*)&Bs[cur][col][kb];
            bfr[nt].f4[1] = *(const float4*)&Bs[cur][col][16 + kb];
        }
        #pragma unroll
        for (int mt = 0; mt < 4; ++mt)
            #pragma unroll
            for (int nt = 0; nt < 2; ++nt)
                acc[mt][nt] = __builtin_amdgcn_wmma_f32_16x16x32_bf16(
                    false, af[mt].v, false, bfr[nt].v,
                    (short)0, acc[mt][nt], false, false);

        stage_fence();
        __syncthreads();
    }

    // ---- epilogue: BN + leaky, pack 8 channels per lane, b128 stores ----
    #pragma unroll
    for (int mt = 0; mt < 4; ++mt) {
        int gmb = Mbase + wm * 64 + mt * 16 + half * 8;
        float sc[8], sh[8];
        #pragma unroll
        for (int r = 0; r < 8; ++r) {
            if (EPI) {
                float g  = bng[gmb + r], bb = bnb[gmb + r];
                float mm = bnm[gmb + r], vv = bnv[gmb + r];
                sc[r] = g * rsqrtf(vv + BN_EPS);
                sh[r] = bb - mm * sc[r];
            } else {
                sc[r] = 1.0f; sh[r] = 0.0f;
            }
        }
        #pragma unroll
        for (int nt = 0; nt < 2; ++nt) {
            int gn = Nbase + wn * 32 + nt * 16 + l15;
            union { uint4 u; __bf16 h[8]; } pk;
            #pragma unroll
            for (int r = 0; r < 8; ++r) {
                float v = acc[mt][nt][r];
                if (EPI) {
                    v = v * sc[r] + sh[r];
                    v = (v >= 0.0f) ? v : LEAKY_SLOPE * v;
                }
                pk.h[r] = (__bf16)v;
            }
            *(uint4*)(Ob + (long)gn * ldo + gmb) = pk.u;
        }
    }
}

// ---------------------------------------------------------------------------
// grid_sample (bilinear, align_corners=True, zeros) + transpose to [n][c]
// grid (32 n-tiles, 8 c-tiles, 8 b), block 1024 = 32(n) x 32(c)
// ---------------------------------------------------------------------------
__global__ __launch_bounds__(1024)
void warp_kernel(const float* __restrict__ src,
                 const float* __restrict__ flow,
                 __bf16* __restrict__ outT)
{
    __shared__ __bf16 tile[32][33];
    int t  = threadIdx.x;
    int tx = t & 31;      // n_local (fast: flow/src access locality)
    int ty = t >> 5;      // c_local
    int n0 = blockIdx.x * 32, c0 = blockIdx.y * 32, b = blockIdx.z;
    int n = n0 + tx, c = c0 + ty;

    const float* fp = flow + ((long)(b << 10) + n) * 2;
    float gx = fp[0], gy = fp[1];
    float xf = (gx + 1.0f) * 15.5f;       // (g+1)*0.5*(32-1)
    float yf = (gy + 1.0f) * 15.5f;
    float x0f = floorf(xf), y0f = floorf(yf);
    int   x0  = (int)x0f,   y0  = (int)y0f;
    float wx1 = xf - x0f, wx0 = 1.0f - wx1;
    float wy1 = yf - y0f, wy0 = 1.0f - wy1;
    const float* im = src + ((long)b * 256 + c) * 1024;
    float v = 0.0f;
    if ((unsigned)x0       < 32u && (unsigned)y0       < 32u) v += im[y0 * 32 + x0]           * (wx0 * wy0);
    if ((unsigned)(x0 + 1) < 32u && (unsigned)y0       < 32u) v += im[y0 * 32 + x0 + 1]       * (wx1 * wy0);
    if ((unsigned)x0       < 32u && (unsigned)(y0 + 1) < 32u) v += im[(y0 + 1) * 32 + x0]     * (wx0 * wy1);
    if ((unsigned)(x0 + 1) < 32u && (unsigned)(y0 + 1) < 32u) v += im[(y0 + 1) * 32 + x0 + 1] * (wx1 * wy1);

    tile[tx][ty] = (__bf16)v;
    __syncthreads();
    // write outT[b][n0+ty][c0+tx], contiguous in c
    outT[(long)b * 256 * 1024 + (long)(n0 + ty) * 256 + (c0 + tx)] = tile[ty][tx];
}

// ---------------------------------------------------------------------------
// conv3 (256 -> 2, 1x1) + bias + flow update (+identity grid at scale 0)
// x2T layout [b][n][256] -> contiguous channel reads. flow f32 [B,32,32,2].
// ---------------------------------------------------------------------------
__global__ void conv3_flow_kernel(const __bf16* __restrict__ x2T,
                                  const float* __restrict__ w3,
                                  const float* __restrict__ b3,
                                  float* __restrict__ flow, int init)
{
    int idx = blockIdx.x * blockDim.x + threadIdx.x;   // b*1024 + n
    if (idx >= 8 * 1024) return;
    int n = idx & 1023;
    int b = idx >> 10;
    const __bf16* xp = x2T + ((long)(b << 10) + n) * 256;
    float s0 = b3[0], s1 = b3[1];
    for (int ci = 0; ci < 256; ++ci) {
        float xv = (float)xp[ci];
        s0 += w3[ci]       * xv;
        s1 += w3[256 + ci] * xv;
    }
    int x = n & 31, y = n >> 5;
    float* o = flow + (long)idx * 2;
    if (init) {
        o[0] = s0 + (x * (2.0f / 31.0f) - 1.0f);
        o[1] = s1 + (y * (2.0f / 31.0f) - 1.0f);
    } else {
        o[0] += s0;
        o[1] += s1;
    }
}

// ---------------------------------------------------------------------------
// Host launcher
// ---------------------------------------------------------------------------
extern "C" void kernel_launch(void* const* d_in, const int* in_sizes, int n_in,
                              void* d_out, int out_size, void* d_ws, size_t ws_size,
                              hipStream_t stream)
{
    (void)in_sizes; (void)n_in; (void)out_size; (void)ws_size;

    const float* src2 = (const float*)d_in[0];
    const float* src3 = (const float*)d_in[1];
    const float* tgt2 = (const float*)d_in[2];
    const float* tgt3 = (const float*)d_in[3];
    const float* w0   = (const float*)d_in[4];
    const float* w1   = (const float*)d_in[5];
    const float* w2   = (const float*)d_in[6];
    const float* w3   = (const float*)d_in[7];
    const float* b3   = (const float*)d_in[8];
    const float* bn0g = (const float*)d_in[9];
    const float* bn0b = (const float*)d_in[10];
    const float* bn0m = (const float*)d_in[11];
    const float* bn0v = (const float*)d_in[12];
    const float* bn1g = (const float*)d_in[13];
    const float* bn1b = (const float*)d_in[14];
    const float* bn1m = (const float*)d_in[15];
    const float* bn1v = (const float*)d_in[16];
    const float* bn2g = (const float*)d_in[17];
    const float* bn2b = (const float*)d_in[18];
    const float* bn2m = (const float*)d_in[19];
    const float* bn2v = (const float*)d_in[20];

    float* flow = (float*)d_out;   // [8,32,32,2]

    // ---- carve workspace ----
    char* wsp = (char*)d_ws;
    auto carve = [&](size_t bytes) -> char* {
        char* p = wsp;
        wsp += (bytes + 255) & ~(size_t)255;
        return p;
    };
    const long LFEAT = 8L * 1024 * 256;                    // per-layer elements
    __bf16* tgtT  = (__bf16*)carve(2 * LFEAT * 2);         // [l][b][s][c] bf16
    __bf16* srcwT = (__bf16*)carve(2 * LFEAT * 2);         // [l][b][s][c] bf16
    __bf16* w0_bf = (__bf16*)carve(1024L * 2048 * 2);      // [co][ci]
    __bf16* w1p   = (__bf16*)carve(512L  * 9216 * 2);      // [co][tap*1024+ci]
    __bf16* w2p   = (__bf16*)carve(256L  * 4608 * 2);      // [co][tap*512+ci]
    __bf16* corrT = (__bf16*)carve(8L * 1024 * 2048 * 2);  // [b][s][2048]
    __bf16* x0T   = (__bf16*)carve(8L * 1024 * 1024 * 2);  // [b][n][1024]
    __bf16* x1T   = (__bf16*)carve(8L * 1024 * 512  * 2);  // [b][n][512]
    __bf16* x2T   = (__bf16*)carve(8L * 1024 * 256  * 2);  // [b][n][256]

    auto cvt = [&](const float* in, __bf16* out, long n) {
        long blocks = (n + 255) / 256;
        if (blocks > 4096) blocks = 4096;
        hipLaunchKernelGGL(cvt_f32_bf16, dim3((unsigned)blocks), dim3(256), 0,
                           stream, in, out, n);
    };

    // features -> transposed bf16 (src copies are scale-0 "warped" inputs)
    hipLaunchKernelGGL(transpose_cvt_f32_bf16, dim3(32, 8, 8), dim3(1024), 0, stream,
                       tgt2, tgtT, 256, 1024);
    hipLaunchKernelGGL(transpose_cvt_f32_bf16, dim3(32, 8, 8), dim3(1024), 0, stream,
                       tgt3, tgtT + LFEAT, 256, 1024);
    hipLaunchKernelGGL(transpose_cvt_f32_bf16, dim3(32, 8, 8), dim3(1024), 0, stream,
                       src2, srcwT, 256, 1024);
    hipLaunchKernelGGL(transpose_cvt_f32_bf16, dim3(32, 8, 8), dim3(1024), 0, stream,
                       src3, srcwT + LFEAT, 256, 1024);

    for (int i = 0; i < 3; ++i) {
        // per-scale weights
        cvt(w0 + (long)i * 1024 * 2048, w0_bf, 1024L * 2048);
        hipLaunchKernelGGL(permute_w_cvt, dim3(4096), dim3(256), 0, stream,
                           w1 + (long)i * 512 * 9216, w1p, 512, 1024);
        hipLaunchKernelGGL(permute_w_cvt, dim3(4096), dim3(256), 0, stream,
                           w2 + (long)i * 256 * 4608, w2p, 256, 512);

        if (i > 0) {
            hipLaunchKernelGGL(warp_kernel, dim3(32, 8, 8), dim3(1024), 0, stream,
                               src2, flow, srcwT);
            hipLaunchKernelGGL(warp_kernel, dim3(32, 8, 8), dim3(1024), 0, stream,
                               src3, flow, srcwT + LFEAT);
        }

        // correlation: corrT[b, s, l*1024 + t] = sum_c tgtT[b,t,c] * srcwT[b,s,c]
        for (int l = 0; l < 2; ++l) {
            hipLaunchKernelGGL((gemm_wmma_kernel<0, 0>), dim3(8, 8, 8), dim3(256), 0, stream,
                tgtT + (long)l * LFEAT, srcwT + (long)l * LFEAT,
                corrT + (long)l * 1024,
                /*K=*/256, /*lda=*/256, /*ldb=*/256, /*ldo=*/2048, /*Ci=*/0,
                (long)1024 * 256, (long)1024 * 256, (long)1024 * 2048,
                nullptr, nullptr, nullptr, nullptr);
        }

        // conv0 (1x1, 2048->1024) + BN0 + leaky
        hipLaunchKernelGGL((gemm_wmma_kernel<0, 1>), dim3(8, 8, 8), dim3(256), 0, stream,
            w0_bf, corrT, x0T,
            2048, 2048, 2048, 1024, 0,
            0L, (long)1024 * 2048, (long)1024 * 1024,
            bn0g + i * 1024, bn0b + i * 1024, bn0m + i * 1024, bn0v + i * 1024);

        // conv1 (3x3, 1024->512) + BN1 + leaky (implicit GEMM, tap-major K)
        hipLaunchKernelGGL((gemm_wmma_kernel<1, 1>), dim3(8, 4, 8), dim3(256), 0, stream,
            w1p, x0T, x1T,
            9216, 9216, 1024, 512, 1024,
            0L, (long)1024 * 1024, (long)1024 * 512,
            bn1g + i * 512, bn1b + i * 512, bn1m + i * 512, bn1v + i * 512);

        // conv2 (3x3, 512->256) + BN2 + leaky
        hipLaunchKernelGGL((gemm_wmma_kernel<1, 1>), dim3(8, 2, 8), dim3(256), 0, stream,
            w2p, x1T, x2T,
            4608, 4608, 512, 256, 512,
            0L, (long)1024 * 512, (long)1024 * 256,
            bn2g + i * 256, bn2b + i * 256, bn2m + i * 256, bn2v + i * 256);

        // conv3 (1x1, 256->2) + bias + flow update
        hipLaunchKernelGGL(conv3_flow_kernel, dim3(32), dim3(256), 0, stream,
            x2T, w3 + (long)i * 512, b3 + (long)i * 2, flow, (i == 0) ? 1 : 0);
    }
}
